// MinkUNetBase_20615843021026
// MI455X (gfx1250) — compile-verified, matrix operand order checked by reading, not run on hardware
//
#include <hip/hip_runtime.h>

// ---------------------------------------------------------------------------
// MinkUNet (dense-voxel emulation) for MI455X / gfx1250.
// Core: implicit-GEMM conv3d on V_WMMA_F32_16X16X32_F16 (f16 in, f32 acc).
// Each wave computes 32 output voxels x 16 output channels (two 16x16 C tiles
// sharing one B operand). The K loop is fully branch-free: concat sources are
// split into uniform chunk ranges, and out-of-range taps read a zeroed scratch
// row via a cndmask'd base pointer instead of EXEC-masked loads.
// ---------------------------------------------------------------------------

typedef __attribute__((ext_vector_type(16))) _Float16 v16h;
typedef __attribute__((ext_vector_type(8)))  _Float16 v8h;
typedef __attribute__((ext_vector_type(8)))  float    v8f;

#define GDIM 64
#define BN_RSQRT 0.99999500003749967f   // 1/sqrt(1 + 1e-5), eval-mode BN

// ---------------------------------------------------------------- voxelize
__global__ void voxelize_scatter_k(const int* __restrict__ gc,
                                   const float* __restrict__ feat,
                                   float* __restrict__ sum,
                                   float* __restrict__ cnt, int n) {
  int i = blockIdx.x * blockDim.x + threadIdx.x;
  if (i >= n) return;
  int v = ((gc[3 * i] * GDIM) + gc[3 * i + 1]) * GDIM + gc[3 * i + 2];
#pragma unroll
  for (int c = 0; c < 6; ++c)
    atomicAdd(&sum[(size_t)v * 6 + c], feat[(size_t)i * 6 + c]);
  atomicAdd(&cnt[v], 1.0f);
}

// Writes a 32-channel zero-padded f16 grid so the stem conv needs no
// elementwise channel guards (channels 6..31 are zero).
__global__ void voxelize_finalize_k(const float* __restrict__ sum,
                                    const float* __restrict__ cnt,
                                    _Float16* __restrict__ vox,
                                    float* __restrict__ mask, int nvox) {
  int v = blockIdx.x * blockDim.x + threadIdx.x;
  if (v >= nvox) return;
  float c = cnt[v];
  float inv = 1.0f / fmaxf(c, 1.0f);
#pragma unroll
  for (int ch = 0; ch < 32; ++ch) {
    float val = (ch < 6) ? sum[(size_t)v * 6 + ch] * inv : 0.0f;
    vox[(size_t)v * 32 + ch] = (_Float16)val;
  }
  mask[v] = (c > 0.0f) ? 1.0f : 0.0f;
}

// ------------------------------------------------------------- mask pooling
__global__ void pool_mask_k(const float* __restrict__ mi,
                            float* __restrict__ mo, int Do) {
  int v = blockIdx.x * blockDim.x + threadIdx.x;
  int n = Do * Do * Do;
  if (v >= n) return;
  int Di = Do * 2;
  int x = v % Do, y = (v / Do) % Do, z = v / (Do * Do);
  float m = 0.0f;
#pragma unroll
  for (int dz = 0; dz < 2; ++dz)
#pragma unroll
    for (int dy = 0; dy < 2; ++dy)
#pragma unroll
      for (int dx = 0; dx < 2; ++dx)
        m = fmaxf(m, mi[(((z * 2 + dz) * Di) + (y * 2 + dy)) * Di + (x * 2 + dx)]);
  mo[v] = m;
}

// ------------------------------------------------- weight repack (f32->f16)
// Blobs of 32 lanes x 16 halves per (tap, K-chunk, N-chunk), matching the ISA
// B-operand layout for V_WMMA_F32_16X16X32_F16:
//   lane = N (mod 16), hi = lane>>4, half h -> K = (h<8 ? hi*8+h : 8+hi*8+h).
// Out-of-range K/N are zero-padded, so conv B loads are two unguarded b128s.
__global__ void repack_w_k(const float* __restrict__ w, _Float16* __restrict__ out,
                           int ntaps, int cin, int cout, int total) {
  int idx = blockIdx.x * blockDim.x + threadIdx.x;
  if (idx >= total) return;
  int KC = (cin + 31) >> 5, NC = (cout + 15) >> 4;
  int h = idx & 15;
  int lane = (idx >> 4) & 31;
  int rest = idx >> 9;
  int nc = rest % NC;
  int kc = (rest / NC) % KC;
  int t = rest / (NC * KC);
  int hi = lane >> 4;
  int n = nc * 16 + (lane & 15);
  int K = kc * 32 + ((h < 8) ? hi * 8 + h : 8 + hi * 8 + h);
  float v = 0.0f;
  if (K < cin && n < cout) v = w[((size_t)t * cin + K) * cout + n];
  out[idx] = (_Float16)v;
}

// ------------------------------------------------------ implicit-GEMM conv3d
template <int TRANSPOSED>
__global__ __launch_bounds__(256) void conv3d_wmma_k(
    const _Float16* __restrict__ in0, const _Float16* __restrict__ in1,
    int c0, int cin,
    const _Float16* __restrict__ wgt,       // repacked (see repack_w_k)
    const float* __restrict__ gamma, const float* __restrict__ beta,
    const float* __restrict__ mask, const _Float16* __restrict__ resid,
    _Float16* __restrict__ out, const _Float16* __restrict__ zrow,
    int Din, int Dout, int cout, int ksize, int stride, int pad, int relu) {
  const int lane = threadIdx.x & 31;
  const int wave = threadIdx.x >> 5;
  const int nvox = Dout * Dout * Dout;
  const int mtile = (blockIdx.x * 8 + wave) * 32;   // 32 voxels per wave
  if (mtile >= nvox) return;                        // wave-uniform
  const int ntile = blockIdx.y * 16;

  const int mrow = lane & 15;
  const int hi = lane >> 4;
  const int KC = (cin + 31) >> 5;
  const int NC = (cout + 15) >> 4;

  const int sh = __builtin_ctz(Dout);               // Dout is a power of two
  const int ov0 = mtile + mrow, ov1 = ov0 + 16;
  const bool ok0 = ov0 < nvox, ok1 = ov1 < nvox;
  const int ox0 = ov0 & (Dout - 1), oy0 = (ov0 >> sh) & (Dout - 1), oz0 = ov0 >> (2 * sh);
  const int ox1 = ov1 & (Dout - 1), oy1 = (ov1 >> sh) & (Dout - 1), oz1 = ov1 >> (2 * sh);

  v8f acc0 = {}, acc1 = {};
  const int ntaps = TRANSPOSED ? 8 : ksize * ksize * ksize;
  const size_t tapStride = (size_t)KC * NC * 512;
  const size_t nBlob = (size_t)(ntile >> 4) * 512;

  int tx = 0, ty = 0, tz = 0;
  for (int t = 0; t < ntaps; ++t) {
    if (TRANSPOSED) { tz = (t >> 2) & 1; ty = (t >> 1) & 1; tx = t & 1; }
    bool valid0 = ok0, valid1 = ok1;
    int iz0, iy0, ix0, iz1, iy1, ix1;
    if (!TRANSPOSED) {
      iz0 = oz0 * stride + tz - pad; iy0 = oy0 * stride + ty - pad; ix0 = ox0 * stride + tx - pad;
      iz1 = oz1 * stride + tz - pad; iy1 = oy1 * stride + ty - pad; ix1 = ox1 * stride + tx - pad;
      valid0 = valid0 && (unsigned)iz0 < (unsigned)Din && (unsigned)iy0 < (unsigned)Din &&
               (unsigned)ix0 < (unsigned)Din;
      valid1 = valid1 && (unsigned)iz1 < (unsigned)Din && (unsigned)iy1 < (unsigned)Din &&
               (unsigned)ix1 < (unsigned)Din;
    } else {                                        // convT k=2 s=2: o = 2*i + tap
      iz0 = oz0 - tz; iy0 = oy0 - ty; ix0 = ox0 - tx;
      iz1 = oz1 - tz; iy1 = oy1 - ty; ix1 = ox1 - tx;
      valid0 = valid0 && iz0 >= 0 && iy0 >= 0 && ix0 >= 0 &&
               !((iz0 | iy0 | ix0) & 1);
      valid1 = valid1 && iz1 >= 0 && iy1 >= 0 && ix1 >= 0 &&
               !((iz1 | iy1 | ix1) & 1);
      iz0 >>= 1; iy0 >>= 1; ix0 >>= 1;
      iz1 >>= 1; iy1 >>= 1; ix1 >>= 1;
    }
    if (!TRANSPOSED) { if (++tx == ksize) { tx = 0; if (++ty == ksize) { ty = 0; ++tz; } } }
    if (!__any(valid0 || valid1)) continue;         // wave-uniform tap skip

    const size_t ivox0 = (size_t)(((iz0 * Din) + iy0) * Din + ix0);
    const size_t ivox1 = (size_t)(((iz1 * Din) + iy1) * Din + ix1);
    const _Float16* wtap = wgt + (size_t)t * tapStride + nBlob;

    if (t + 1 < ntaps)                              // -> global_prefetch_b8
      __builtin_prefetch(wtap + tapStride + lane * 16, 0, 0);

    // Branch-free K loop over one uniform source range. Invalid rows read a
    // zeroed scratch row: address cndmask only, EXEC untouched.
    auto krange = [&](const _Float16* src, int cw, int kciBeg, int kciEnd) {
      if (kciBeg >= kciEnd) return;
      const _Float16* pr0 = valid0 ? src + ivox0 * cw : zrow;
      const _Float16* pr1 = valid1 ? src + ivox1 * cw : zrow;
      const _Float16* wl = wtap + (size_t)kciBeg * NC * 512 + lane * 16;
      int cb = hi * 8;
      for (int kci = kciBeg; kci < kciEnd; ++kci, cb += 32) {
        union { v16h v; v8h g[2]; } A0, A1, B;
        A0.g[0] = *(const v8h*)(pr0 + cb);
        A0.g[1] = *(const v8h*)(pr0 + cb + 16);
        A1.g[0] = *(const v8h*)(pr1 + cb);
        A1.g[1] = *(const v8h*)(pr1 + cb + 16);
        B.g[0] = *(const v8h*)(wl);
        B.g[1] = *(const v8h*)(wl + 8);
        wl += (size_t)NC * 512;
        acc0 = __builtin_amdgcn_wmma_f32_16x16x32_f16(
            false, A0.v, false, B.v, (short)0, acc0, false, false);
        acc1 = __builtin_amdgcn_wmma_f32_16x16x32_f16(
            false, A1.v, false, B.v, (short)0, acc1, false, false);
      }
    };
    // All concat split points (c0) are multiples of 32, so each 32-wide
    // K-chunk lives entirely in one source.
    krange(in0, c0, 0, c0 >> 5);
    if (in1) krange(in1, cin - c0, c0 >> 5, KC);
  }

  // ---- epilogue: BN -> mask -> residual -> ReLU -> f16 store ----
  const int ch = ntile + mrow;                      // C layout: lane%16 = N
  const bool chOk = ch < cout;
  float scale = 1.0f, bias = 0.0f;
  if (gamma != nullptr && chOk) { scale = gamma[ch] * BN_RSQRT; bias = beta[ch]; }
#pragma unroll
  for (int r = 0; r < 8; ++r) {                     // C layout: VGPR r -> M = r + 8*hi
    int vr0 = mtile + r + hi * 8;
    int vr1 = vr0 + 16;
    if (vr0 < nvox && chOk) {
      float v = acc0[r];
      if (gamma) v = v * scale + bias;
      if (mask) v *= mask[vr0];
      if (resid) v += (float)resid[(size_t)vr0 * cout + ch];
      if (relu) v = fmaxf(v, 0.0f);
      out[(size_t)vr0 * cout + ch] = (_Float16)v;
    }
    if (vr1 < nvox && chOk) {
      float v = acc1[r];
      if (gamma) v = v * scale + bias;
      if (mask) v *= mask[vr1];
      if (resid) v += (float)resid[(size_t)vr1 * cout + ch];
      if (relu) v = fmaxf(v, 0.0f);
      out[(size_t)vr1 * cout + ch] = (_Float16)v;
    }
  }
}

// ----------------------------------------------- final 1x1 head + point gather
__global__ void final_head_k(const int* __restrict__ gc,
                             const _Float16* __restrict__ x,
                             const float* __restrict__ fw,
                             const float* __restrict__ fb,
                             float* __restrict__ out, int n) {
  int i = blockIdx.x * blockDim.x + threadIdx.x;
  if (i >= n) return;
  int v = ((gc[3 * i] * GDIM) + gc[3 * i + 1]) * GDIM + gc[3 * i + 2];
  const _Float16* xr = x + (size_t)v * 96;
#pragma unroll
  for (int o = 0; o < 20; ++o) {
    float s = fb[o];
    for (int c = 0; c < 96; ++c) s += (float)xr[c] * fw[c * 20 + o];
    out[(size_t)i * 20 + o] = s;
  }
}

// ===========================================================================
extern "C" void kernel_launch(void* const* d_in, const int* in_sizes, int n_in,
                              void* d_out, int out_size, void* d_ws, size_t ws_size,
                              hipStream_t stream) {
  (void)n_in; (void)out_size; (void)ws_size;
  const int* grid_coord = (const int*)d_in[0];
  const float* feat = (const float*)d_in[1];
  // d_in[2] = offset (single batch) -- unused
  const int N = in_sizes[0] / 3;

  // ---- parse params (reference-file insertion order, recursively) ----
  int p = 3;
  auto F = [&]() { return (const float*)d_in[p++]; };
  struct BN { const float *g, *b; };
  auto rBN = [&]() { BN r; r.g = F(); r.b = F(); return r; };
  struct Blk { const float* w1; BN bn1; const float* w2; BN bn2;
               const float* dsw; BN dsbn; bool ds; };
  auto rBlk = [&](bool ds) {
    Blk k{}; k.w1 = F(); k.bn1 = rBN(); k.w2 = F(); k.bn2 = rBN();
    if (ds) { k.dsw = F(); k.dsbn = rBN(); }
    k.ds = ds; return k;
  };
  const float* conv0w = F(); BN bn0 = rBN();
  const float* conv1w = F(); BN bn1n = rBN();
  Blk B1[2] = { rBlk(false), rBlk(false) };
  const float* conv2w = F(); BN bn2n = rBN();
  Blk B2[2] = { rBlk(true), rBlk(false) };
  const float* conv3w = F(); BN bn3n = rBN();
  Blk B3[2] = { rBlk(true), rBlk(false) };
  const float* conv4w = F(); BN bn4n = rBN();
  Blk B4[2] = { rBlk(true), rBlk(false) };
  const float* ctr4 = F(); BN btr4 = rBN();
  Blk B5[2] = { rBlk(true), rBlk(false) };
  const float* ctr5 = F(); BN btr5 = rBN();
  Blk B6[2] = { rBlk(true), rBlk(false) };
  const float* ctr6 = F(); BN btr6 = rBN();
  Blk B7[2] = { rBlk(true), rBlk(false) };
  const float* ctr7 = F(); BN btr7 = rBN();
  Blk B8[2] = { rBlk(true), rBlk(false) };
  const float* final_w = F(); const float* final_b = F();

  // ---- deterministic bump allocator over d_ws ----
  char* base = (char*)d_ws;
  size_t off = 0;
  auto alloc = [&](size_t bytes) -> void* {
    off = (off + 255) & ~(size_t)255;
    void* r = base + off; off += bytes; return r;
  };
  const size_t NV1 = 64 * 64 * 64, NV2 = 32 * 32 * 32, NV4 = 16 * 16 * 16,
               NV8 = 8 * 8 * 8, NV16 = 4 * 4 * 4;
  float*    sum6  = (float*)alloc(NV1 * 6 * 4);
  float*    cnt   = (float*)alloc(NV1 * 4);
  _Float16* vox   = (_Float16*)alloc(NV1 * 32 * 2);    // zero-padded to 32 ch
  _Float16* zrow  = (_Float16*)alloc(1024 * 2);        // zeroed scratch row
  float*    m1    = (float*)alloc(NV1 * 4);
  float*    m2    = (float*)alloc(NV2 * 4);
  float*    m4    = (float*)alloc(NV4 * 4);
  float*    m8    = (float*)alloc(NV8 * 4);
  float*    m16   = (float*)alloc(NV16 * 4);
  _Float16* outP1 = (_Float16*)alloc(NV1 * 32 * 2);
  _Float16* b1s   = (_Float16*)alloc(NV2 * 32 * 2);
  _Float16* b2s   = (_Float16*)alloc(NV4 * 64 * 2);
  _Float16* b3s   = (_Float16*)alloc(NV8 * 128 * 2);
  _Float16* pool[3];
  for (int i = 0; i < 3; ++i) pool[i] = (_Float16*)alloc(NV1 * 96 * 2);
  int xi = 0;

  // ---- device-side weight repack into WMMA B-operand layout ----
  auto rp = [&](const float* w, int ntaps, int cin, int cout) -> const _Float16* {
    int KC = (cin + 31) >> 5, NC = (cout + 15) >> 4;
    int total = ntaps * KC * NC * 512;
    _Float16* d = (_Float16*)alloc((size_t)total * 2);
    repack_w_k<<<dim3((total + 255) / 256), dim3(256), 0, stream>>>(
        w, d, ntaps, cin, cout, total);
    return d;
  };
  auto conv = [&](const _Float16* in0, const _Float16* in1, int c0, int cin,
                  const _Float16* w, BN bn, bool useBN, const float* mask,
                  const _Float16* resid, _Float16* out, int Din, int Dout,
                  int cout, int ks, int st, int pad, int tr, int relu) {
    int nvox = Dout * Dout * Dout;
    dim3 g((nvox + 255) / 256, (cout + 15) / 16);
    const float* gp = useBN ? bn.g : nullptr;
    const float* bp = useBN ? bn.b : nullptr;
    if (tr)
      conv3d_wmma_k<1><<<g, 256, 0, stream>>>(in0, in1, c0, cin, w, gp, bp, mask,
          resid, out, zrow, Din, Dout, cout, ks, st, pad, relu);
    else
      conv3d_wmma_k<0><<<g, 256, 0, stream>>>(in0, in1, c0, cin, w, gp, bp, mask,
          resid, out, zrow, Din, Dout, cout, ks, st, pad, relu);
  };
  // basic residual block on the 3-buffer rotation; optional concat second input
  auto runBlock = [&](const Blk& K, int D, const float* mask, int cin, int cout,
                      const _Float16* in1, int c0) {
    _Float16* x = pool[xi];
    _Float16* t = pool[(xi + 1) % 3];
    _Float16* o = pool[(xi + 2) % 3];
    const _Float16* W1 = rp(K.w1, 27, cin, cout);
    const _Float16* W2 = rp(K.w2, 27, cout, cout);
    if (K.ds) {
      const _Float16* DW = rp(K.dsw, 1, cin, cout);
      conv(x, in1, c0, cin, DW, K.dsbn, true, mask, nullptr, o, D, D, cout, 1, 1, 0, 0, 0);
      conv(x, in1, c0, cin, W1, K.bn1, true, mask, nullptr, t, D, D, cout, 3, 1, 1, 0, 1);
      conv(t, nullptr, cout, cout, W2, K.bn2, true, mask, o, x, D, D, cout, 3, 1, 1, 0, 1);
      // x stays at pool[xi]
    } else {
      conv(x, nullptr, cin, cin, W1, K.bn1, true, mask, nullptr, t, D, D, cout, 3, 1, 1, 0, 1);
      conv(t, nullptr, cout, cout, W2, K.bn2, true, mask, x, o, D, D, cout, 3, 1, 1, 0, 1);
      xi = (xi + 2) % 3;
    }
  };
  auto down = [&](const float* w, BN bn, const float* mask, int Din, int Dout,
                  int cin, int cout) {
    const _Float16* W = rp(w, 8, cin, cout);
    _Float16* o = pool[(xi + 1) % 3];
    conv(pool[xi], nullptr, cin, cin, W, bn, true, mask, nullptr, o,
         Din, Dout, cout, 2, 2, 0, 0, 1);
    xi = (xi + 1) % 3;
  };
  auto up = [&](const float* w, BN bn, const float* mask, int Din, int Dout,
                int cin, int cout) {
    const _Float16* W = rp(w, 8, cin, cout);
    _Float16* o = pool[(xi + 1) % 3];
    conv(pool[xi], nullptr, cin, cin, W, bn, true, mask, nullptr, o,
         Din, Dout, cout, 2, 2, 0, 1, 1);
    xi = (xi + 1) % 3;
  };

  // ---- voxelize + occupancy masks ----
  hipMemsetAsync(sum6, 0, NV1 * 6 * 4, stream);
  hipMemsetAsync(cnt, 0, NV1 * 4, stream);
  hipMemsetAsync(zrow, 0, 1024 * 2, stream);
  voxelize_scatter_k<<<dim3((N + 255) / 256), dim3(256), 0, stream>>>(
      grid_coord, feat, sum6, cnt, N);
  voxelize_finalize_k<<<dim3(((int)NV1 + 255) / 256), dim3(256), 0, stream>>>(
      sum6, cnt, vox, m1, (int)NV1);
  pool_mask_k<<<dim3(((int)NV2 + 255) / 256), dim3(256), 0, stream>>>(m1, m2, 32);
  pool_mask_k<<<dim3(((int)NV4 + 255) / 256), dim3(256), 0, stream>>>(m2, m4, 16);
  pool_mask_k<<<dim3(((int)NV8 + 255) / 256), dim3(256), 0, stream>>>(m4, m8, 8);
  pool_mask_k<<<dim3(((int)NV16 + 255) / 256), dim3(256), 0, stream>>>(m8, m16, 4);

  // ---- stem (k=5, SAME); input zero-padded to 32 ch, weights K-padded ----
  conv(vox, nullptr, 32, 32, rp(conv0w, 125, 6, 32), bn0, true, m1, nullptr,
       outP1, 64, 64, 32, 5, 1, 2, 0, 1);

  // ---- encoder ----
  conv(outP1, nullptr, 32, 32, rp(conv1w, 8, 32, 32), bn1n, true, m2, nullptr,
       pool[0], 64, 32, 32, 2, 2, 0, 0, 1);
  xi = 0;
  runBlock(B1[0], 32, m2, 32, 32, nullptr, 32);
  runBlock(B1[1], 32, m2, 32, 32, nullptr, 32);
  hipMemcpyAsync(b1s, pool[xi], NV2 * 32 * 2, hipMemcpyDeviceToDevice, stream);

  down(conv2w, bn2n, m4, 32, 16, 32, 32);
  runBlock(B2[0], 16, m4, 32, 64, nullptr, 32);
  runBlock(B2[1], 16, m4, 64, 64, nullptr, 64);
  hipMemcpyAsync(b2s, pool[xi], NV4 * 64 * 2, hipMemcpyDeviceToDevice, stream);

  down(conv3w, bn3n, m8, 16, 8, 64, 64);
  runBlock(B3[0], 8, m8, 64, 128, nullptr, 64);
  runBlock(B3[1], 8, m8, 128, 128, nullptr, 128);
  hipMemcpyAsync(b3s, pool[xi], NV8 * 128 * 2, hipMemcpyDeviceToDevice, stream);

  down(conv4w, bn4n, m16, 8, 4, 128, 128);
  runBlock(B4[0], 4, m16, 128, 256, nullptr, 128);
  runBlock(B4[1], 4, m16, 256, 256, nullptr, 256);

  // ---- decoder (concat via dual-input convs; skips never materialized) ----
  up(ctr4, btr4, m8, 4, 8, 256, 256);
  runBlock(B5[0], 8, m8, 256 + 128, 256, b3s, 256);
  runBlock(B5[1], 8, m8, 256, 256, nullptr, 256);

  up(ctr5, btr5, m4, 8, 16, 256, 128);
  runBlock(B6[0], 16, m4, 128 + 64, 128, b2s, 128);
  runBlock(B6[1], 16, m4, 128, 128, nullptr, 128);

  up(ctr6, btr6, m2, 16, 32, 128, 96);
  runBlock(B7[0], 32, m2, 96 + 32, 96, b1s, 96);
  runBlock(B7[1], 32, m2, 96, 96, nullptr, 96);

  up(ctr7, btr7, m1, 32, 64, 96, 96);
  runBlock(B8[0], 64, m1, 96 + 32, 96, outP1, 96);
  runBlock(B8[1], 64, m1, 96, 96, nullptr, 96);

  // ---- final 1x1 head + gather back to points ----
  final_head_k<<<dim3((N + 255) / 256), dim3(256), 0, stream>>>(
      grid_coord, pool[xi], final_w, final_b, (float*)d_out, N);
}
